// WaterPredictor_7009386627598
// MI455X (gfx1250) — matrix operand order repaired and checked by reading, not compiled, not run on hardware
//
#include <hip/hip_runtime.h>
#include <hip/hip_bf16.h>

// ---------------------------------------------------------------------------
// EGT water-network forward for gfx1250 (CDNA5, wave32, WMMA f16->f32,
// async global->LDS DMA for weight preloads and the edge gather).
//
// Input index map (setup_inputs() insertion order, params flattened
// depth-first in dict/list insertion order):
//   0 x[N,26]  1 edge_attr[E,15]  2 edge_index[2,E] (int32)
//   3 ng.Wih[192,2] 4 ng.Whh[192,64] 5 ng.bih 6 ng.bhh
//   7/8 node_gru_ln g/b   9/10 node_elev W/b   11 node_type_emb[10,32]
//   12/13 node_fusion W[64,128]/b   14/15 node_fusion_ln g/b
//   16 eg.Wih[192] 17 eg.Whh[192,64] 18 eg.bih 19 eg.bhh
//   20/21 edge_gru_ln g/b   22/23 edge_static W[64,3]/b
//   24/25 edge_fusion W[64,128]/b   26/27 edge_fusion_ln g/b
//   layer l (base = 28 + 24*l):
//     +0 qW +1 qb +2 kW +3 kb +4 vW +5 vb +6 oW +7 ob
//     +8 ebW[4,64] +9 ebb[4] +10/11 ln1 g/b +12/13 ln2 g/b
//     +14 ffn1W[128,64] +15 ffn1b +16 ffn2W[64,128] +17 ffn2b
//     +18/19 ln_e g/b +20 eu1W[64,192] +21 eu1b +22 eu2W[64,64] +23 eu2b
//   76/77 node_pred1 W/b  78/79 node_pred2 W[1,64]/b[1]
//   80/81 edge_pred1 W/b  82/83 edge_pred2 W/b
// Output: float [N + E]  (pressure then flow)
// ---------------------------------------------------------------------------

typedef __attribute__((ext_vector_type(16))) _Float16 v16h;
typedef __attribute__((ext_vector_type(8)))  float    v8f;

#define DEV __device__ __forceinline__

DEV int imin(int a, int b) { return a < b ? a : b; }
DEV int imax(int a, int b) { return a > b ? a : b; }

DEV v8f v8zero() {
  v8f z = {0.f, 0.f, 0.f, 0.f, 0.f, 0.f, 0.f, 0.f};
  return z;
}

DEV float sigmoid_f(float x) { return 1.0f / (1.0f + expf(-x)); }
DEV float gelu_f(float x) { return 0.5f * x * (1.0f + erff(x * 0.70710678118654752f)); }

// Low 32 bits of a generic pointer into __shared__ are the LDS byte offset
// (this is exactly what the compiler feeds ds_* address operands).
DEV unsigned lds_off_u32(const void* p) {
  return (unsigned)(unsigned long long)p;
}

// Async global->LDS DMA (GLOBAL_LOAD_ASYNC_TO_LDS_B128, GV mode, ASYNCcnt).
DEV void async_copy_b128(const void* gsrc, unsigned lds_off) {
  asm volatile("global_load_async_to_lds_b128 %0, %1, off"
               :
               : "v"(lds_off), "v"(gsrc)
               : "memory");
}
DEV void wait_async0() { asm volatile("s_wait_asynccnt 0x0" ::: "memory"); }

// ---- WMMA fragment helpers (CDNA5 16x16x32 f16 layouts) -------------------
// A (16xK tile): rowk must point at element k0 of this lane's row (m=lane&15).
DEV v16h a_frag(const _Float16* rowk, int lane) {
  const int kb = (lane >> 4) << 3;  // lanes 0-15: K 0..7/16..23 ; 16-31: 8..15/24..31
  v16h a;
#pragma unroll
  for (int e = 0; e < 8; ++e) a[e] = rowk[kb + e];
#pragma unroll
  for (int e = 0; e < 8; ++e) a[e + 8] = rowk[16 + kb + e];
  return a;
}

// B tile from row-major weight W[nout][K] (y = x @ W^T): B[k][n] = W[n][k].
DEV v16h b_frag(const _Float16* W, int ld, int n0, int k0, int lane) {
  const int nn = lane & 15;
  const int kb = (lane >> 4) << 3;  // lanes 0-15: K 0..7/16..23 ; 16-31: 8..15/24..31
  const _Float16* r = W + (size_t)(n0 + nn) * ld + k0 + kb;
  v16h b;
#pragma unroll
  for (int e = 0; e < 8; ++e) b[e] = r[e];
#pragma unroll
  for (int e = 0; e < 8; ++e) b[e + 8] = r[16 + e];
  return b;
}

DEV v8f wmma32(const v16h& a, const v16h& b, const v8f& c) {
  return __builtin_amdgcn_wmma_f32_16x16x32_f16(false, a, false, b, (short)0, c,
                                                false, false);
}

// LayerNorm over a 16x64 D-tile set (acc[4]; cols 16j + (lane&15), rows e).
// Cross-lane shuffle reduction within each 16-lane half (wave32).
DEV void ln_tiles4(v8f* acc, const float* g, const float* b, int lane) {
  const int n = lane & 15;
#pragma unroll
  for (int e = 0; e < 8; ++e) {
    float s = acc[0][e] + acc[1][e] + acc[2][e] + acc[3][e];
    s += __shfl_xor(s, 1);
    s += __shfl_xor(s, 2);
    s += __shfl_xor(s, 4);
    s += __shfl_xor(s, 8);
    const float mean = s * 0.015625f;
    float q = 0.f;
#pragma unroll
    for (int j = 0; j < 4; ++j) {
      const float d = acc[j][e] - mean;
      q += d * d;
    }
    q += __shfl_xor(q, 1);
    q += __shfl_xor(q, 2);
    q += __shfl_xor(q, 4);
    q += __shfl_xor(q, 8);
    const float inv = rsqrtf(q * 0.015625f + 1e-5f);
#pragma unroll
    for (int j = 0; j < 4; ++j)
      acc[j][e] = (acc[j][e] - mean) * inv * g[16 * j + n] + b[16 * j + n];
  }
}

// Order-preserving float <-> uint for atomic segment-max.
DEV unsigned int f32_ord(float f) {
  unsigned int u = __float_as_uint(f);
  return (u & 0x80000000u) ? ~u : (u | 0x80000000u);
}
DEV float ord_f32(unsigned int u) {
  u = (u & 0x80000000u) ? (u & 0x7fffffffu) : ~u;
  return __uint_as_float(u);
}

// ---------------------------------------------------------------------------
__global__ void cvt_f32_to_f16_kernel(const float* __restrict__ s,
                                      _Float16* __restrict__ d, int n) {
  const int i = blockIdx.x * 256 + threadIdx.x;
  if (i < n) d[i] = (_Float16)s[i];
}

// ---------------------------------------------------------------------------
// Node lifting: GRU(2->64, 12 steps) in LDS + LN + fused concat GEMM + LN + GELU
__global__ __launch_bounds__(256) void node_lift_kernel(
    const float* __restrict__ x, const _Float16* __restrict__ Whh16,
    const float* __restrict__ Wih, const float* __restrict__ bih,
    const float* __restrict__ bhh, const float* __restrict__ gln_g,
    const float* __restrict__ gln_b, const float* __restrict__ elevW,
    const float* __restrict__ elevb, const float* __restrict__ temb,
    const _Float16* __restrict__ fusW16, const float* __restrict__ fusb,
    const float* __restrict__ fln_g, const float* __restrict__ fln_b,
    float* __restrict__ h_node, _Float16* __restrict__ h_node_h, int Nn) {
  __shared__ _Float16 sh_h[128][64];     // 16 KB
  __shared__ _Float16 sh_whh[192 * 64];  // 24 KB
  const int tid = threadIdx.x;
  const int lane = tid & 31;
  const int wave = tid >> 5;
  const int row0 = blockIdx.x * 128;

  // Async DMA Whh into LDS (1536 x 16B chunks), overlapped with h init.
  for (int c = tid; c < 1536; c += 256)
    async_copy_b128(Whh16 + (size_t)c * 8, lds_off_u32(&sh_whh[c * 8]));
  for (int i = tid; i < 128 * 64; i += 256) (&sh_h[0][0])[i] = (_Float16)0.0f;

  const int wrow = wave * 16;
  const int n = lane & 15;
  const int mb = (lane >> 4) << 3;

  // per-lane gate weights: index g3*4+j covers (gate g3, dim 16j+n)
  float wf0[12], wf1[12], bi[12], bh[12];
#pragma unroll
  for (int g3 = 0; g3 < 3; ++g3)
#pragma unroll
    for (int j = 0; j < 4; ++j) {
      const int rr = g3 * 64 + 16 * j + n;
      wf0[g3 * 4 + j] = Wih[rr * 2 + 0];
      wf1[g3 * 4 + j] = Wih[rr * 2 + 1];
      bi[g3 * 4 + j] = bih[rr];
      bh[g3 * 4 + j] = bhh[rr];
    }

  wait_async0();
  __syncthreads();

  for (int t = 0; t < 12; ++t) {
    v8f acc[12];
#pragma unroll
    for (int j = 0; j < 12; ++j) acc[j] = v8zero();
#pragma unroll
    for (int kt = 0; kt < 2; ++kt) {
      const int k0 = kt * 32;
      v16h a = a_frag(&sh_h[wrow][0] + (lane & 15) * 64 + k0, lane);
#pragma unroll
      for (int j = 0; j < 12; ++j) {
        v16h b = b_frag(sh_whh, 64, 16 * j, k0, lane);
        acc[j] = wmma32(a, b, acc[j]);
      }
    }
#pragma unroll
    for (int e = 0; e < 8; ++e) {
      const int r = wrow + mb + e;
      const int node = row0 + r;
      float x0 = 0.0f, x1 = 0.0f;
      if (node < Nn) {
        x0 = x[(size_t)node * 26 + t];
        x1 = x[(size_t)node * 26 + 12 + t];
      }
#pragma unroll
      for (int j = 0; j < 4; ++j) {
        const float hr = acc[j][e] + bh[j];
        const float hz = acc[4 + j][e] + bh[4 + j];
        const float hn = acc[8 + j][e] + bh[8 + j];
        const float ir = x0 * wf0[j] + x1 * wf1[j] + bi[j];
        const float iz = x0 * wf0[4 + j] + x1 * wf1[4 + j] + bi[4 + j];
        const float in_ = x0 * wf0[8 + j] + x1 * wf1[8 + j] + bi[8 + j];
        const float rg = sigmoid_f(ir + hr);
        const float zg = sigmoid_f(iz + hz);
        const float ng = tanhf(in_ + rg * hn);
        const float ho = (float)sh_h[r][16 * j + n];
        sh_h[r][16 * j + n] = (_Float16)((1.0f - zg) * ng + zg * ho);
      }
    }
  }

  // GRU output LayerNorm, keep normalized h_dyn in LDS (f16)
  {
    v8f hacc[4];
#pragma unroll
    for (int j = 0; j < 4; ++j)
#pragma unroll
      for (int e = 0; e < 8; ++e)
        hacc[j][e] = (float)sh_h[wrow + mb + e][16 * j + n];
    ln_tiles4(hacc, gln_g, gln_b, lane);
#pragma unroll
    for (int j = 0; j < 4; ++j)
#pragma unroll
      for (int e = 0; e < 8; ++e)
        sh_h[wrow + mb + e][16 * j + n] = (_Float16)hacc[j][e];
  }

  // fusion: [elev(32) | type(32) | h_dyn(64)] @ W^T + b -> LN -> GELU
  const int myrow = wrow + (lane & 15);
  const int mynode = imin(row0 + myrow, Nn - 1);
  const float myelev = x[(size_t)mynode * 26 + 24];
  int myti = (int)x[(size_t)mynode * 26 + 25];
  myti = imax(0, imin(9, myti));

  v8f facc[4];
#pragma unroll
  for (int j = 0; j < 4; ++j) facc[j] = v8zero();
#pragma unroll
  for (int kt = 0; kt < 4; ++kt) {
    const int k0 = kt * 32;
    const int kb = (lane >> 4) << 3;
    v16h a;
#pragma unroll
    for (int e2 = 0; e2 < 16; ++e2) {
      const int kk = k0 + ((e2 < 8) ? (kb + e2) : (16 + kb + (e2 - 8)));
      float v;
      if (kk < 32)
        v = myelev * elevW[kk] + elevb[kk];
      else if (kk < 64)
        v = temb[myti * 32 + (kk - 32)];
      else
        v = (float)sh_h[myrow][kk - 64];
      a[e2] = (_Float16)v;
    }
#pragma unroll
    for (int j = 0; j < 4; ++j) {
      v16h b = b_frag(fusW16, 128, 16 * j, k0, lane);
      facc[j] = wmma32(a, b, facc[j]);
    }
  }
#pragma unroll
  for (int j = 0; j < 4; ++j)
#pragma unroll
    for (int e = 0; e < 8; ++e) facc[j][e] += fusb[16 * j + n];
  ln_tiles4(facc, fln_g, fln_b, lane);
#pragma unroll
  for (int j = 0; j < 4; ++j)
#pragma unroll
    for (int e = 0; e < 8; ++e) {
      const int node = row0 + wrow + mb + e;
      if (node < Nn) {
        const float v = gelu_f(facc[j][e]);
        h_node[(size_t)node * 64 + 16 * j + n] = v;
        h_node_h[(size_t)node * 64 + 16 * j + n] = (_Float16)v;
      }
    }
}

// ---------------------------------------------------------------------------
// Edge lifting: GRU(1->64) + LN + [static(64)|h_dyn(64)] fusion + LN + GELU
__global__ __launch_bounds__(256) void edge_lift_kernel(
    const float* __restrict__ ea, const _Float16* __restrict__ Whh16,
    const float* __restrict__ Wih, const float* __restrict__ bih,
    const float* __restrict__ bhh, const float* __restrict__ gln_g,
    const float* __restrict__ gln_b, const float* __restrict__ statW,
    const float* __restrict__ statb, const _Float16* __restrict__ fusW16,
    const float* __restrict__ fusb, const float* __restrict__ fln_g,
    const float* __restrict__ fln_b, float* __restrict__ h_edge,
    _Float16* __restrict__ h_edge_h, int Ee) {
  __shared__ _Float16 sh_h[128][64];
  __shared__ _Float16 sh_whh[192 * 64];
  const int tid = threadIdx.x;
  const int lane = tid & 31;
  const int wave = tid >> 5;
  const int row0 = blockIdx.x * 128;

  for (int c = tid; c < 1536; c += 256)
    async_copy_b128(Whh16 + (size_t)c * 8, lds_off_u32(&sh_whh[c * 8]));
  for (int i = tid; i < 128 * 64; i += 256) (&sh_h[0][0])[i] = (_Float16)0.0f;

  const int wrow = wave * 16;
  const int n = lane & 15;
  const int mb = (lane >> 4) << 3;

  float wf[12], bi[12], bh[12];
#pragma unroll
  for (int g3 = 0; g3 < 3; ++g3)
#pragma unroll
    for (int j = 0; j < 4; ++j) {
      const int rr = g3 * 64 + 16 * j + n;
      wf[g3 * 4 + j] = Wih[rr];
      bi[g3 * 4 + j] = bih[rr];
      bh[g3 * 4 + j] = bhh[rr];
    }

  wait_async0();
  __syncthreads();

  for (int t = 0; t < 12; ++t) {
    v8f acc[12];
#pragma unroll
    for (int j = 0; j < 12; ++j) acc[j] = v8zero();
#pragma unroll
    for (int kt = 0; kt < 2; ++kt) {
      const int k0 = kt * 32;
      v16h a = a_frag(&sh_h[wrow][0] + (lane & 15) * 64 + k0, lane);
#pragma unroll
      for (int j = 0; j < 12; ++j) {
        v16h b = b_frag(sh_whh, 64, 16 * j, k0, lane);
        acc[j] = wmma32(a, b, acc[j]);
      }
    }
#pragma unroll
    for (int e = 0; e < 8; ++e) {
      const int r = wrow + mb + e;
      const int edge = row0 + r;
      float x0 = 0.0f;
      if (edge < Ee) x0 = ea[(size_t)edge * 15 + t];
#pragma unroll
      for (int j = 0; j < 4; ++j) {
        const float hr = acc[j][e] + bh[j];
        const float hz = acc[4 + j][e] + bh[4 + j];
        const float hn = acc[8 + j][e] + bh[8 + j];
        const float ir = x0 * wf[j] + bi[j];
        const float iz = x0 * wf[4 + j] + bi[4 + j];
        const float in_ = x0 * wf[8 + j] + bi[8 + j];
        const float rg = sigmoid_f(ir + hr);
        const float zg = sigmoid_f(iz + hz);
        const float ng = tanhf(in_ + rg * hn);
        const float ho = (float)sh_h[r][16 * j + n];
        sh_h[r][16 * j + n] = (_Float16)((1.0f - zg) * ng + zg * ho);
      }
    }
  }

  {
    v8f hacc[4];
#pragma unroll
    for (int j = 0; j < 4; ++j)
#pragma unroll
      for (int e = 0; e < 8; ++e)
        hacc[j][e] = (float)sh_h[wrow + mb + e][16 * j + n];
    ln_tiles4(hacc, gln_g, gln_b, lane);
#pragma unroll
    for (int j = 0; j < 4; ++j)
#pragma unroll
      for (int e = 0; e < 8; ++e)
        sh_h[wrow + mb + e][16 * j + n] = (_Float16)hacc[j][e];
  }

  const int myrow = wrow + (lane & 15);
  const int myedge = imin(row0 + myrow, Ee - 1);
  const float s0 = ea[(size_t)myedge * 15 + 12];
  const float s1 = ea[(size_t)myedge * 15 + 13];
  const float s2 = ea[(size_t)myedge * 15 + 14];

  v8f facc[4];
#pragma unroll
  for (int j = 0; j < 4; ++j) facc[j] = v8zero();
#pragma unroll
  for (int kt = 0; kt < 4; ++kt) {
    const int k0 = kt * 32;
    const int kb = (lane >> 4) << 3;
    v16h a;
#pragma unroll
    for (int e2 = 0; e2 < 16; ++e2) {
      const int kk = k0 + ((e2 < 8) ? (kb + e2) : (16 + kb + (e2 - 8)));
      float v;
      if (kk < 64)
        v = s0 * statW[kk * 3 + 0] + s1 * statW[kk * 3 + 1] +
            s2 * statW[kk * 3 + 2] + statb[kk];
      else
        v = (float)sh_h[myrow][kk - 64];
      a[e2] = (_Float16)v;
    }
#pragma unroll
    for (int j = 0; j < 4; ++j) {
      v16h b = b_frag(fusW16, 128, 16 * j, k0, lane);
      facc[j] = wmma32(a, b, facc[j]);
    }
  }
#pragma unroll
  for (int j = 0; j < 4; ++j)
#pragma unroll
    for (int e = 0; e < 8; ++e) facc[j][e] += fusb[16 * j + n];
  ln_tiles4(facc, fln_g, fln_b, lane);
#pragma unroll
  for (int j = 0; j < 4; ++j)
#pragma unroll
    for (int e = 0; e < 8; ++e) {
      const int edge = row0 + wrow + mb + e;
      if (edge < Ee) {
        const float v = gelu_f(facc[j][e]);
        h_edge[(size_t)edge * 64 + 16 * j + n] = v;
        h_edge_h[(size_t)edge * 64 + 16 * j + n] = (_Float16)v;
      }
    }
}

// ---------------------------------------------------------------------------
__global__ __launch_bounds__(256) void qkv_kernel(
    const _Float16* __restrict__ hN16, const _Float16* __restrict__ qW,
    const _Float16* __restrict__ kW, const _Float16* __restrict__ vW,
    const float* __restrict__ qb, const float* __restrict__ kb2,
    const float* __restrict__ vb, _Float16* __restrict__ qkv, int Nn) {
  const int tid = threadIdx.x, lane = tid & 31, wave = tid >> 5;
  const int row0 = blockIdx.x * 128 + wave * 16;
  const int n = lane & 15, mb = (lane >> 4) << 3;
  const int am = imin(row0 + (lane & 15), Nn - 1);
  v8f acc[12];
#pragma unroll
  for (int j = 0; j < 12; ++j) acc[j] = v8zero();
#pragma unroll
  for (int kt = 0; kt < 2; ++kt) {
    const int k0 = kt * 32;
    v16h a = a_frag(hN16 + (size_t)am * 64 + k0, lane);
#pragma unroll
    for (int j = 0; j < 12; ++j) {
      const _Float16* W = (j < 4) ? qW : (j < 8) ? kW : vW;
      v16h b = b_frag(W, 64, 16 * (j & 3), k0, lane);
      acc[j] = wmma32(a, b, acc[j]);
    }
  }
#pragma unroll
  for (int j = 0; j < 12; ++j) {
    const float* bb = (j < 4) ? qb : (j < 8) ? kb2 : vb;
#pragma unroll
    for (int e = 0; e < 8; ++e) {
      const int row = row0 + mb + e;
      if (row < Nn)
        qkv[(size_t)row * 192 + 16 * j + n] =
            (_Float16)(acc[j][e] + bb[16 * (j & 3) + n]);
    }
  }
}

// ---------------------------------------------------------------------------
__global__ void attn_init_kernel(unsigned int* __restrict__ smax,
                                 float* __restrict__ ssum,
                                 float* __restrict__ attn, int Nn) {
  const int i = blockIdx.x * 256 + threadIdx.x;
  if (i < Nn * 4) {
    smax[i] = 0u;  // ordered encoding of -inf
    ssum[i] = 0.f;
  }
  if (i < Nn * 64) attn[i] = 0.f;
}

__global__ __launch_bounds__(256) void score_kernel(
    const int* __restrict__ ei, const _Float16* __restrict__ qkv,
    const _Float16* __restrict__ hE16, const float* __restrict__ ebW,
    const float* __restrict__ ebb, float* __restrict__ scoreb,
    unsigned int* __restrict__ smax, int Ee) {
  const int e = blockIdx.x * 256 + threadIdx.x;
  if (e >= Ee) return;
  const int src = ei[e];
  const int dst = ei[Ee + e];
  const _Float16* q = qkv + (size_t)dst * 192;
  const _Float16* k = qkv + (size_t)src * 192 + 64;
  const _Float16* he = hE16 + (size_t)e * 64;
  __builtin_prefetch((const void*)q, 0, 3);
  __builtin_prefetch((const void*)k, 0, 3);
#pragma unroll
  for (int h = 0; h < 4; ++h) {
    float s = 0.f;
#pragma unroll
    for (int i = 0; i < 16; ++i) s += (float)q[16 * h + i] * (float)k[16 * h + i];
    s *= 0.25f;  // DH^-0.5
    float bb = ebb[h];
#pragma unroll
    for (int d = 0; d < 64; ++d) bb += (float)he[d] * ebW[h * 64 + d];
    s += bb;
    scoreb[(size_t)e * 4 + h] = s;
    atomicMax(&smax[(size_t)dst * 4 + h], f32_ord(s));
  }
}

__global__ __launch_bounds__(256) void attn_acc_kernel(
    const int* __restrict__ ei, const _Float16* __restrict__ qkv,
    const float* __restrict__ scoreb, const unsigned int* __restrict__ smax,
    float* __restrict__ ssum, float* __restrict__ attn, int Ee) {
  const int e = blockIdx.x * 256 + threadIdx.x;
  if (e >= Ee) return;
  const int src = ei[e];
  const int dst = ei[Ee + e];
  const _Float16* v = qkv + (size_t)src * 192 + 128;
  __builtin_prefetch((const void*)v, 0, 3);
  float ex[4];
#pragma unroll
  for (int h = 0; h < 4; ++h) {
    const float m = ord_f32(smax[(size_t)dst * 4 + h]);
    ex[h] = expf(scoreb[(size_t)e * 4 + h] - m);
    atomicAdd(&ssum[(size_t)dst * 4 + h], ex[h]);
  }
  float* o = attn + (size_t)dst * 64;
#pragma unroll
  for (int d = 0; d < 64; ++d) atomicAdd(&o[d], (float)v[d] * ex[d >> 4]);
}

// ---------------------------------------------------------------------------
// Node update: o-GEMM (softmax normalization folded into A), ln1, FFN, ln2.
__global__ __launch_bounds__(256) void node_update_kernel(
    const float* hN, const float* __restrict__ attn,
    const float* __restrict__ ssum, const _Float16* __restrict__ oW,
    const float* __restrict__ ob, const float* __restrict__ ln1g,
    const float* __restrict__ ln1b, const _Float16* __restrict__ f1W,
    const float* __restrict__ f1b, const _Float16* __restrict__ f2W,
    const float* __restrict__ f2b, const float* __restrict__ ln2g,
    const float* __restrict__ ln2b, float* hNo, _Float16* __restrict__ hNo16,
    int Nn) {
  __shared__ _Float16 sh1[8][16 * 64];   // 16 KB
  __shared__ _Float16 sh2[8][16 * 128];  // 32 KB
  const int tid = threadIdx.x, lane = tid & 31, wave = tid >> 5;
  const int row0 = blockIdx.x * 128 + wave * 16;
  const int n = lane & 15, mb = (lane >> 4) << 3;

  // (1) attention output / ssum, then o-projection
  const int arow = imin(row0 + (lane & 15), Nn - 1);
  const float* ar = attn + (size_t)arow * 64;
  float inv_s[4];
#pragma unroll
  for (int h = 0; h < 4; ++h) {
    const float s = ssum[(size_t)arow * 4 + h];
    inv_s[h] = (s > 0.f) ? (1.f / s) : 0.f;
  }
  v8f acc[4];
#pragma unroll
  for (int j = 0; j < 4; ++j) acc[j] = v8zero();
#pragma unroll
  for (int kt = 0; kt < 2; ++kt) {
    const int k0 = kt * 32;
    const int kb = (lane >> 4) << 3;
    v16h a;
#pragma unroll
    for (int e2 = 0; e2 < 16; ++e2) {
      const int kk = k0 + ((e2 < 8) ? (kb + e2) : (16 + kb + (e2 - 8)));
      a[e2] = (_Float16)(ar[kk] * inv_s[kk >> 4]);
    }
#pragma unroll
    for (int j = 0; j < 4; ++j) {
      v16h b = b_frag(oW, 64, 16 * j, k0, lane);
      acc[j] = wmma32(a, b, acc[j]);
    }
  }
#pragma unroll
  for (int j = 0; j < 4; ++j)
#pragma unroll
    for (int e = 0; e < 8; ++e) {
      const int row = imin(row0 + mb + e, Nn - 1);
      acc[j][e] += ob[16 * j + n] + hN[(size_t)row * 64 + 16 * j + n];
    }
  ln_tiles4(acc, ln1g, ln1b, lane);

  _Float16* h1 = sh1[wave];
#pragma unroll
  for (int j = 0; j < 4; ++j)
#pragma unroll
    for (int e = 0; e < 8; ++e)
      h1[(mb + e) * 64 + 16 * j + n] = (_Float16)acc[j][e];

  // (2) ffn1 64->128 + GELU
  v8f f1[8];
#pragma unroll
  for (int j = 0; j < 8; ++j) f1[j] = v8zero();
#pragma unroll
  for (int kt = 0; kt < 2; ++kt) {
    v16h a = a_frag(h1 + (lane & 15) * 64 + kt * 32, lane);
#pragma unroll
    for (int j = 0; j < 8; ++j) {
      v16h b = b_frag(f1W, 64, 16 * j, kt * 32, lane);
      f1[j] = wmma32(a, b, f1[j]);
    }
  }
  _Float16* actb = sh2[wave];
#pragma unroll
  for (int j = 0; j < 8; ++j)
#pragma unroll
    for (int e = 0; e < 8; ++e)
      actb[(mb + e) * 128 + 16 * j + n] =
          (_Float16)gelu_f(f1[j][e] + f1b[16 * j + n]);

  // (3) ffn2 128->64, residual (acc holds ln1 output), ln2
  v8f f2[4];
#pragma unroll
  for (int j = 0; j < 4; ++j) f2[j] = v8zero();
#pragma unroll
  for (int kt = 0; kt < 4; ++kt) {
    v16h a = a_frag(actb + (lane & 15) * 128 + kt * 32, lane);
#pragma unroll
    for (int j = 0; j < 4; ++j) {
      v16h b = b_frag(f2W, 128, 16 * j, kt * 32, lane);
      f2[j] = wmma32(a, b, f2[j]);
    }
  }
#pragma unroll
  for (int j = 0; j < 4; ++j)
#pragma unroll
    for (int e = 0; e < 8; ++e) acc[j][e] += f2[j][e] + f2b[16 * j + n];
  ln_tiles4(acc, ln2g, ln2b, lane);
#pragma unroll
  for (int j = 0; j < 4; ++j)
#pragma unroll
    for (int e = 0; e < 8; ++e) {
      const int row = row0 + mb + e;
      if (row < Nn) {
        const float v = acc[j][e];
        hNo[(size_t)row * 64 + 16 * j + n] = v;
        hNo16[(size_t)row * 64 + 16 * j + n] = (_Float16)v;
      }
    }
}

// ---------------------------------------------------------------------------
// Edge update: async-gather [h_src | h_dst | h_edge] (192) into LDS,
// eu1 -> GELU -> eu2 -> ln_e.
__global__ __launch_bounds__(256) void edge_update_kernel(
    const int* __restrict__ ei, const _Float16* __restrict__ hN16,
    const float* hEf, const _Float16* hE16, const _Float16* __restrict__ e1W,
    const float* __restrict__ e1b, const _Float16* __restrict__ e2W,
    const float* __restrict__ e2b, const float* __restrict__ lneg,
    const float* __restrict__ lneb, float* hEfo, _Float16* hE16o, int Ee) {
  __shared__ _Float16 shA[8][16 * 192];  // 48 KB
  const int tid = threadIdx.x, lane = tid & 31, wave = tid >> 5;
  const int row0 = blockIdx.x * 128 + wave * 16;
  const int n = lane & 15, mb = (lane >> 4) << 3;
  _Float16* A = shA[wave];

  // Async DMA gather: 2 lanes per row, 96 cols (12 x 16B chunks) each.
  // Row layout: cols [0,64) = h_node[src], [64,128) = h_node[dst],
  // [128,192) = h_edge[e]; all 16B-aligned segments.
  {
    const int r = lane >> 1;
    const int cb = (lane & 1) * 96;
    const int e = imin(row0 + r, Ee - 1);
    const int s = ei[e];
    const int d = ei[Ee + e];
    const _Float16* seg0 = hN16 + (size_t)s * 64;
    const _Float16* seg1 = hN16 + (size_t)d * 64;
    const _Float16* seg2 = hE16 + (size_t)e * 64;
#pragma unroll
    for (int c = 0; c < 12; ++c) {
      const int col = cb + 8 * c;  // multiple of 8 halfs = 16B
      const int sel = col >> 6;    // 0,1,2
      const _Float16* g =
          ((sel == 0) ? seg0 : (sel == 1) ? seg1 : seg2) + (col & 63);
      async_copy_b128(g, lds_off_u32(&A[r * 192 + col]));
    }
  }
  wait_async0();  // this wave reads only tiles it loaded itself

  // eu1: K=192, N=64
  v8f g1[4];
#pragma unroll
  for (int j = 0; j < 4; ++j) g1[j] = v8zero();
#pragma unroll
  for (int kt = 0; kt < 6; ++kt) {
    v16h a = a_frag(A + (lane & 15) * 192 + kt * 32, lane);
#pragma unroll
    for (int j = 0; j < 4; ++j) {
      v16h b = b_frag(e1W, 192, 16 * j, kt * 32, lane);
      g1[j] = wmma32(a, b, g1[j]);
    }
  }
  // GELU -> act, reusing the (fully consumed) A region
  _Float16* act = A;
#pragma unroll
  for (int j = 0; j < 4; ++j)
#pragma unroll
    for (int e = 0; e < 8; ++e)
      act[(mb + e) * 64 + 16 * j + n] =
          (_Float16)gelu_f(g1[j][e] + e1b[16 * j + n]);

  // eu2: K=64, N=64
  v8f g2[4];
#pragma unroll
  for (int j = 0; j < 4; ++j) g2[j] = v8zero();
#pragma unroll
  for (int kt = 0; kt < 2; ++kt) {
    v16h a = a_frag(act + (lane & 15) * 64 + kt * 32, lane);
#pragma unroll
    for (int j = 0; j < 4; ++j) {
      v16h b = b_frag(e2W, 64, 16 * j, kt * 32, lane);
      g2[j] = wmma32(a, b, g2[j]);
    }
  }
#pragma unroll
  for (int j = 0; j < 4; ++j)
#pragma unroll
    for (int e = 0; e < 8; ++e) {
      const int ee = imin(row0 + mb + e, Ee - 1);
      g2[j][e] += e2b[16 * j + n] + hEf[(size_t)ee * 64 + 16 * j + n];
    }
  ln_tiles4(g2, lneg, lneb, lane);
#pragma unroll
  for (int j = 0; j < 4; ++j)
#pragma unroll
    for (int e = 0; e < 8; ++e) {
      const int ee = row0 + mb + e;
      if (ee < Ee) {
        const float v = g2[j][e];
        hEfo[(size_t)ee * 64 + 16 * j + n] = v;
        hE16o[(size_t)ee * 64 + 16 * j + n] = (_Float16)v;
      }
    }
}

// ---------------------------------------------------------------------------
// Prediction head: pred1 (WMMA) + GELU + pred2 dot (shuffle reduce)
__global__ __launch_bounds__(256) void head_kernel(
    const _Float16* __restrict__ hin16, const _Float16* __restrict__ p1W,
    const float* __restrict__ p1b, const float* __restrict__ p2W,
    const float* __restrict__ p2b, float* __restrict__ out, int M) {
  const int tid = threadIdx.x, lane = tid & 31, wave = tid >> 5;
  const int row0 = blockIdx.x * 128 + wave * 16;
  const int n = lane & 15, mb = (lane >> 4) << 3;
  const int am = imin(row0 + (lane & 15), M - 1);
  v8f acc[4];
#pragma unroll
  for (int j = 0; j < 4; ++j) acc[j] = v8zero();
#pragma unroll
  for (int kt = 0; kt < 2; ++kt) {
    v16h a = a_frag(hin16 + (size_t)am * 64 + kt * 32, lane);
#pragma unroll
    for (int j = 0; j < 4; ++j) {
      v16h b = b_frag(p1W, 64, 16 * j, kt * 32, lane);
      acc[j] = wmma32(a, b, acc[j]);
    }
  }
#pragma unroll
  for (int e = 0; e < 8; ++e) {
    float s = 0.f;
#pragma unroll
    for (int j = 0; j < 4; ++j)
      s += gelu_f(acc[j][e] + p1b[16 * j + n]) * p2W[16 * j + n];
    s += __shfl_xor(s, 1);
    s += __shfl_xor(s, 2);
    s += __shfl_xor(s, 4);
    s += __shfl_xor(s, 8);
    const int row = row0 + mb + e;
    if (n == 0 && row < M) out[row] = s + p2b[0];
  }
}

// ---------------------------------------------------------------------------
extern "C" void kernel_launch(void* const* d_in, const int* in_sizes, int n_in,
                              void* d_out, int out_size, void* d_ws,
                              size_t ws_size, hipStream_t stream) {
  (void)n_in;
  (void)out_size;
  (void)ws_size;
  const int N = in_sizes[0] / 26;
  const int E = in_sizes[1] / 15;
  const int* ei = (const int*)d_in[2];
  auto P = [&](int i) { return (const float*)d_in[i]; };

  char* ws = (char*)d_ws;
  size_t off = 0;
  auto alloc = [&](size_t bytes) -> void* {
    void* p = (void*)(ws + off);
    off += (bytes + 255) & ~(size_t)255;
    return p;
  };
  auto cvt = [&](int idx, int nelem) -> _Float16* {
    _Float16* dst = (_Float16*)alloc((size_t)nelem * sizeof(_Float16));
    cvt_f32_to_f16_kernel<<<(nelem + 255) / 256, 256, 0, stream>>>(P(idx), dst,
                                                                   nelem);
    return dst;
  };

  // f16 weight mirrors
  _Float16* W_ngWhh = cvt(4, 192 * 64);
  _Float16* W_nfus = cvt(12, 64 * 128);
  _Float16* W_egWhh = cvt(17, 192 * 64);
  _Float16* W_efus = cvt(24, 64 * 128);
  _Float16 *W_q[2], *W_k[2], *W_v[2], *W_o[2], *W_f1[2], *W_f2[2], *W_e1[2],
      *W_e2[2];
  for (int l = 0; l < 2; ++l) {
    const int b = 28 + 24 * l;
    W_q[l] = cvt(b + 0, 64 * 64);
    W_k[l] = cvt(b + 2, 64 * 64);
    W_v[l] = cvt(b + 4, 64 * 64);
    W_o[l] = cvt(b + 6, 64 * 64);
    W_f1[l] = cvt(b + 14, 128 * 64);
    W_f2[l] = cvt(b + 16, 64 * 128);
    W_e1[l] = cvt(b + 20, 64 * 192);
    W_e2[l] = cvt(b + 22, 64 * 64);
  }
  _Float16* W_np1 = cvt(76, 64 * 64);
  _Float16* W_ep1 = cvt(80, 64 * 64);

  // activations / scratch
  float* h_node = (float*)alloc((size_t)N * 64 * 4);
  _Float16* h_node_h = (_Float16*)alloc((size_t)N * 64 * 2);
  float* h_edge = (float*)alloc((size_t)E * 64 * 4);
  _Float16* h_edge_h = (_Float16*)alloc((size_t)E * 64 * 2);
  _Float16* qkvb = (_Float16*)alloc((size_t)N * 192 * 2);
  float* scoreb = (float*)alloc((size_t)E * 4 * 4);
  unsigned int* smax = (unsigned int*)alloc((size_t)N * 4 * 4);
  float* ssum = (float*)alloc((size_t)N * 4 * 4);
  float* attnb = (float*)alloc((size_t)N * 64 * 4);

  const int nbN = (N + 127) / 128;
  const int nbE = (E + 127) / 128;

  node_lift_kernel<<<nbN, 256, 0, stream>>>(
      P(0), W_ngWhh, P(3), P(5), P(6), P(7), P(8), P(9), P(10), P(11), W_nfus,
      P(13), P(14), P(15), h_node, h_node_h, N);
  edge_lift_kernel<<<nbE, 256, 0, stream>>>(
      P(1), W_egWhh, P(16), P(18), P(19), P(20), P(21), P(22), P(23), W_efus,
      P(25), P(26), P(27), h_edge, h_edge_h, E);

  for (int l = 0; l < 2; ++l) {
    const int b = 28 + 24 * l;
    qkv_kernel<<<nbN, 256, 0, stream>>>(h_node_h, W_q[l], W_k[l], W_v[l],
                                        P(b + 1), P(b + 3), P(b + 5), qkvb, N);
    attn_init_kernel<<<(N * 64 + 255) / 256, 256, 0, stream>>>(smax, ssum,
                                                               attnb, N);
    score_kernel<<<(E + 255) / 256, 256, 0, stream>>>(
        ei, qkvb, h_edge_h, P(b + 8), P(b + 9), scoreb, smax, E);
    attn_acc_kernel<<<(E + 255) / 256, 256, 0, stream>>>(ei, qkvb, scoreb,
                                                         smax, ssum, attnb, E);
    node_update_kernel<<<nbN, 256, 0, stream>>>(
        h_node, attnb, ssum, W_o[l], P(b + 7), P(b + 10), P(b + 11), W_f1[l],
        P(b + 15), W_f2[l], P(b + 17), P(b + 12), P(b + 13), h_node, h_node_h,
        N);
    edge_update_kernel<<<nbE, 256, 0, stream>>>(
        ei, h_node_h, h_edge, h_edge_h, W_e1[l], P(b + 21), W_e2[l], P(b + 23),
        P(b + 18), P(b + 19), h_edge, h_edge_h, E);
  }

  head_kernel<<<nbN, 256, 0, stream>>>(h_node_h, W_np1, P(77), P(78), P(79),
                                       (float*)d_out, N);
  head_kernel<<<nbE, 256, 0, stream>>>(h_edge_h, W_ep1, P(81), P(82), P(83),
                                       (float*)d_out + N, E);
}
